// ECG_XNOR_48309792145691
// MI455X (gfx1250) — 8-bit matrix operand order repaired and checked by reading, compiled, not run on hardware
//
#include <hip/hip_runtime.h>
#include <hip/hip_bf16.h>

typedef int v8i __attribute__((ext_vector_type(8)));

#define BN_EPS 1e-5f

constexpr int c_CIN[7]   = {1, 8, 12, 32, 64, 64, 64};
constexpr int c_COUT[7]  = {8, 12, 32, 64, 64, 64, 72};
constexpr int c_KW[7]    = {16, 12, 9, 7, 5, 3, 3};
constexpr int c_STR[7]   = {2, 2, 1, 1, 1, 1, 1};
constexpr int c_PAD[7]   = {7, 5, 4, 3, 2, 1, 1};
constexpr int c_LOUT[7]  = {1800, 224, 111, 54, 26, 13, 6};
constexpr int c_POOL[7]  = {8, 4, 5, 4, 2, 2, 2};
constexpr int c_PST[7]   = {4, 2, 2, 2, 2, 2, 2};
constexpr int c_LPOOL[7] = {449, 111, 54, 26, 13, 6, 3};
constexpr int c_RS[7]    = {3614, 459, 119, 60, 30, 15, 8};

struct Params {
  const float* x;
  const float* w[7];
  const float* b[7];
  const float* g[7];
  const float* be[7];
  const float* m[7];
  const float* v[7];
  const float* lw1;
  const float* lb1;
  const float* lw2;
  const float* lb2;
  float* out;
};

template <int PB>
__device__ __forceinline__ void run_wmma_block(
    const Params& P, int tid, int wid, int lane, int half, int l16,
    signed char* __restrict__ actIn, signed char* __restrict__ actOut,
    signed char* __restrict__ s_a, signed char* __restrict__ s_w,
    short* __restrict__ s_y2, float* __restrict__ s_bias,
    float* __restrict__ s_bnscale, float* __restrict__ s_bnoff,
    signed char* __restrict__ s_feat) {
  constexpr int  cout    = c_COUT[PB];
  constexpr int  Kw      = c_KW[PB];
  constexpr int  stride  = c_STR[PB];
  constexpr int  rs      = c_RS[PB];
  constexpr int  lout    = c_LOUT[PB];
  constexpr int  KK      = c_CIN[PB] * Kw;
  constexpr int  KKp     = (KK + 63) & ~63;
  constexpr int  coutP   = (cout + 15) & ~15;
  constexpr int  pool    = c_POOL[PB];
  constexpr int  pst     = c_PST[PB];
  constexpr int  lpool   = c_LPOOL[PB];
  constexpr int  Mtiles  = (lout + 15) / 16;
  constexpr int  Ntiles  = coutP / 16;
  constexpr int  Ksteps  = KKp / 64;
  constexpr int  LoutPad = Mtiles * 16;
  constexpr bool LAST    = (PB == 6);
  constexpr int  nextPad = LAST ? 0 : c_PAD[PB + 1];
  constexpr int  nextRs  = LAST ? 0 : c_RS[PB + 1];

  if (!LAST) {
    for (int i = tid; i < 4608; i += 256) actOut[i] = 0;
  }
  const float* wsrc = P.w[PB];
  for (int i = tid; i < coutP * KKp; i += 256) {
    int oc = i / KKp;
    int kk = i - oc * KKp;
    signed char val = 0;
    if (oc < cout && kk < KK)
      val = (wsrc[oc * KK + kk] >= 0.0f) ? (signed char)1 : (signed char)-1;
    s_w[i] = val;
  }
  if (tid < cout) s_bias[tid] = P.b[PB][tid];
  if (!LAST) {
    if (tid < cout) {
      float rr = rsqrtf(P.v[PB + 1][tid] + BN_EPS);
      float ss = P.g[PB + 1][tid] * rr;
      s_bnscale[tid] = ss;
      s_bnoff[tid]   = P.be[PB + 1][tid] - P.m[PB + 1][tid] * ss;
    }
  }
  __syncthreads();

  for (int l = wid; l < LoutPad; l += 8) {
    const bool lv = l < lout;
    signed char* dst = s_a + l * KKp;
    const signed char* src = actIn + l * stride;
    for (int kk4 = lane * 4; kk4 < KKp; kk4 += 128) {
      int w32 = 0;
#pragma unroll
      for (int j = 0; j < 4; ++j) {
        int kk = kk4 + j;
        int c  = kk / Kw;
        int by = 0;
        if (lv && kk < KK) by = (int)src[c * rs + (kk - c * Kw)];
        w32 |= (by & 0xFF) << (8 * j);
      }
      *(int*)(dst + kk4) = w32;
    }
  }
  __syncthreads();

  for (int t = wid; t < Mtiles * Ntiles; t += 8) {
    int mt = t % Mtiles;
    int nt = t / Mtiles;
    const signed char* arow = s_a + (mt * 16 + l16) * KKp;
    const signed char* wrow = s_w + (nt * 16 + l16) * KKp;

    auto loadA = [&](int kt, v8i& a) {
#pragma unroll
      for (int vv = 0; vv < 8; ++vv) {
        int Ka = ((vv >> 1) << 4) + ((vv & 1) << 2) + (half << 3);
        a[vv] = *(const int*)(arow + kt * 64 + Ka);
      }
    };
    auto loadB = [&](int kt, v8i& b) {
#pragma unroll
      for (int vv = 0; vv < 8; ++vv) {
        int Kb = ((vv >> 1) << 4) + ((vv & 1) << 2) + (half << 3);
        b[vv] = *(const int*)(wrow + kt * 64 + Kb);
      }
    };

    v8i acc[4];
#pragma unroll
    for (int q = 0; q < 4; ++q) acc[q] = (v8i){0, 0, 0, 0, 0, 0, 0, 0};
    v8i af[2], bf[2];
    loadA(0, af[0]);
    loadB(0, bf[0]);
#pragma unroll
    for (int kt = 0; kt < Ksteps; ++kt) {
      if (kt + 1 < Ksteps) {
        loadA(kt + 1, af[(kt + 1) & 1]);
        loadB(kt + 1, bf[(kt + 1) & 1]);
      }
      acc[kt & 3] = __builtin_amdgcn_wmma_i32_16x16x64_iu8(
          true, af[kt & 1], true, bf[kt & 1], acc[kt & 3], false, false);
    }
#pragma unroll
    for (int r = 0; r < 8; ++r) {
      int mm = mt * 16 + r + (half << 3);
      int oc = nt * 16 + l16;
      int sum = acc[0][r];
      if (Ksteps > 1) sum += acc[1][r];
      if (Ksteps > 2) sum += acc[2][r];
      if (Ksteps > 3) sum += acc[3][r];
      if (mm < lout && oc < cout) s_y2[oc * lout + mm] = (short)sum;
    }
  }
  __syncthreads();

  for (int idx = tid; idx < cout * lpool; idx += 256) {
    int oc = idx / lpool;
    int p  = idx - oc * lpool;
    short mv = s_y2[oc * lout + p * pst];
#pragma unroll
    for (int j = 1; j < pool; ++j) {
      short sv = s_y2[oc * lout + p * pst + j];
      mv = sv > mv ? sv : mv;
    }
    float yv = (float)mv + s_bias[oc];
    if (!LAST) {
      actOut[oc * nextRs + nextPad + p] =
          (s_bnscale[oc] * yv + s_bnoff[oc] >= 0.0f) ? (signed char)1 : (signed char)-1;
    } else {
      s_feat[oc * 3 + p] = (yv >= 0.0f) ? (signed char)1 : (signed char)-1;
    }
  }
  __syncthreads();
}

__global__ __launch_bounds__(256) void ecg_xnor_fused(Params P) {
  const int sample = blockIdx.x;
  const int tid  = threadIdx.x;
  const int wid  = tid >> 5;
  const int lane = tid & 31;
  const int half = lane >> 4;
  const int l16  = lane & 15;

  __shared__ int s_big_i[7200];
  __shared__ short s_y2[3584];
  __shared__ int s_w_i[5120];
  __shared__ signed char s_actA[4608];
  __shared__ signed char s_actB[4608];
  __shared__ float s_bias[80];
  __shared__ float s_bnscale[80];
  __shared__ float s_bnoff[80];
  __shared__ signed char s_feat[216];
  __shared__ signed char s_fc1[64];

  signed char* s_w  = (signed char*)s_w_i;
  signed char* s_a  = (signed char*)s_big_i;
  short*       s_y0 = (short*)s_big_i;

  {
    const float r0  = rsqrtf(P.v[0][0] + BN_EPS);
    const float sc0 = P.g[0][0] * r0;
    const float of0 = P.be[0][0] - P.m[0][0] * sc0;

    for (int i = tid; i < 4608; i += 256) { s_actA[i] = 0; s_actB[i] = 0; }
    __syncthreads();

    const float* xin = P.x + (size_t)sample * 3600;
    for (int i = tid; i < 3600; i += 256) {
      float xv = xin[i];
      s_actA[7 + i] = (sc0 * xv + of0 >= 0.0f) ? (signed char)1 : (signed char)-1;
    }
    for (int i = tid; i < 128; i += 256)
      s_w[i] = (P.w[0][i] >= 0.0f) ? (signed char)1 : (signed char)-1;
    if (tid < 8) s_bias[tid] = P.b[0][tid];
    if (tid < 8) {
      float rr = rsqrtf(P.v[1][tid] + BN_EPS);
      float ss = P.g[1][tid] * rr;
      s_bnscale[tid] = ss;
      s_bnoff[tid]   = P.be[1][tid] - P.m[1][tid] * ss;
    }
    __syncthreads();

    for (int idx = tid; idx < 8 * 1800; idx += 256) {
      int oc = idx / 1800;
      int l  = idx - oc * 1800;
      int base = l * 2;
      int sum = 0;
#pragma unroll
      for (int k = 0; k < 16; ++k)
        sum += (int)s_w[oc * 16 + k] * (int)s_actA[base + k];
      s_y0[oc * 1800 + l] = (short)sum;
    }
    __syncthreads();

    for (int idx = tid; idx < 8 * 449; idx += 256) {
      int oc = idx / 449;
      int p  = idx - oc * 449;
      short mv = s_y0[oc * 1800 + p * 4];
#pragma unroll
      for (int j = 1; j < 8; ++j) {
        short sv = s_y0[oc * 1800 + p * 4 + j];
        mv = sv > mv ? sv : mv;
      }
      float yv = (float)mv + s_bias[oc];
      s_actB[oc * 459 + 5 + p] =
          (s_bnscale[oc] * yv + s_bnoff[oc] >= 0.0f) ? (signed char)1 : (signed char)-1;
    }
    __syncthreads();
  }

  run_wmma_block<1>(P, tid, wid, lane, half, l16, s_actB, s_actA, s_a, s_w, s_y2,
                    s_bias, s_bnscale, s_bnoff, s_feat);
  run_wmma_block<2>(P, tid, wid, lane, half, l16, s_actA, s_actB, s_a, s_w, s_y2,
                    s_bias, s_bnscale, s_bnoff, s_feat);
  run_wmma_block<3>(P, tid, wid, lane, half, l16, s_actB, s_actA, s_a, s_w, s_y2,
                    s_bias, s_bnscale, s_bnoff, s_feat);
  run_wmma_block<4>(P, tid, wid, lane, half, l16, s_actA, s_actB, s_a, s_w, s_y2,
                    s_bias, s_bnscale, s_bnoff, s_feat);
  run_wmma_block<5>(P, tid, wid, lane, half, l16, s_actB, s_actA, s_a, s_w, s_y2,
                    s_bias, s_bnscale, s_bnoff, s_feat);
  run_wmma_block<6>(P, tid, wid, lane, half, l16, s_actA, s_actB, s_a, s_w, s_y2,
                    s_bias, s_bnscale, s_bnoff, s_feat);

  if (tid < 64) {
    const float* wr = P.lw1 + tid * 216;
    int acc = 0;
    for (int i = 0; i < 216; ++i) {
      int wv = (wr[i] >= 0.0f) ? 1 : -1;
      acc += wv * (int)s_feat[i];
    }
    float val = (float)acc + P.lb1[tid];
    s_fc1[tid] = (val >= 0.0f) ? (signed char)1 : (signed char)-1;
  }
  __syncthreads();

  if (tid < 17) {
    const float* wr = P.lw2 + tid * 64;
    int acc = 0;
    for (int i = 0; i < 64; ++i) {
      int wv = (wr[i] >= 0.0f) ? 1 : -1;
      acc += wv * (int)s_fc1[i];
    }
    P.out[(size_t)sample * 17 + tid] = (float)acc + P.lb2[tid];
  }
}

extern "C" void kernel_launch(void* const* d_in, const int* in_sizes, int n_in,
                              void* d_out, int out_size, void* d_ws, size_t ws_size,
                              hipStream_t stream) {
  (void)n_in; (void)out_size; (void)d_ws; (void)ws_size;
  Params P;
  P.x = (const float*)d_in[0];
  for (int i = 0; i < 7; ++i) {
    P.w[i]  = (const float*)d_in[1 + 6 * i + 0];
    P.b[i]  = (const float*)d_in[1 + 6 * i + 1];
    P.g[i]  = (const float*)d_in[1 + 6 * i + 2];
    P.be[i] = (const float*)d_in[1 + 6 * i + 3];
    P.m[i]  = (const float*)d_in[1 + 6 * i + 4];
    P.v[i]  = (const float*)d_in[1 + 6 * i + 5];
  }
  P.lw1 = (const float*)d_in[43];
  P.lb1 = (const float*)d_in[44];
  P.lw2 = (const float*)d_in[45];
  P.lb2 = (const float*)d_in[46];
  P.out = (float*)d_out;

  const int batch = in_sizes[0] / 3600;
  dim3 grid((unsigned)batch), block(256);
  hipLaunchKernelGGL(ecg_xnor_fused, grid, block, 0, stream, P);
}
